// OursLayer2_51213190037998
// MI455X (gfx1250) — matrix operand order repaired and checked by reading, not compiled, not run on hardware
//
#include <hip/hip_runtime.h>
#include <hip/hip_bf16.h>
#include <math.h>

typedef __hip_bfloat16 bf16_t;
typedef __attribute__((ext_vector_type(16))) __bf16 v16bf;
typedef __attribute__((ext_vector_type(8)))  __bf16 v8bf;
typedef __attribute__((ext_vector_type(8)))  float  v8f;

#define N_NODES 8192
#define M_NODES 4096
#define F_IN    256
#define F_OUT   128
#define B_SRC   2048
#define NEGV    (-9e15f)

// register-blocking factors: each wave computes (16*MT) x (16*NT) output
#define MT 2
#define NT 4

static __device__ __forceinline__ __bf16 f2b(float x) {
  __hip_bfloat16 h = __float2bfloat16(x);
  return *reinterpret_cast<__bf16*>(&h);
}

// ---------------- elementwise prep ----------------

__global__ void cast_f32_bf16(const float* __restrict__ src, bf16_t* __restrict__ dst, int n) {
  int i = blockIdx.x * 256 + threadIdx.x;
  if (i < n) dst[i] = __float2bfloat16(src[i]);
}

// W [K,Nc] row-major -> Wt [Nc,K] row-major (bf16)
__global__ void transpose_cast(const float* __restrict__ W, bf16_t* __restrict__ Wt, int K, int Nc) {
  int i = blockIdx.x * 256 + threadIdx.x;
  if (i < K * Nc) {
    int k = i / Nc, n = i - k * Nc;
    Wt[(size_t)n * K + k] = __float2bfloat16(W[i]);
  }
}

__global__ void zero_f32(float* __restrict__ p, int n) {
  int i = blockIdx.x * 256 + threadIdx.x;
  if (i < n) p[i] = 0.f;
}

// ---------------- generic bf16 WMMA GEMM, 32x64 per wave ----------------
// D[M,N] = A[M,K] (row-major bf16, stride LD) x Bt[N,K]^T (row-major bf16, stride LD)
//          + (Cin ? Cin : 0)
// LD is a compile-time constant so all tile offsets fold into load immediates
// (2 base address pairs instead of 6 row pointers -> no spills, pipelined loads).
// One wave per (16*MT)x(16*NT) macro-tile. grid = (N/(16*NT), M/(16*MT)).
// act: 0 = none, 2 = ELU (final output).
template <int LD>
__global__ void gemm_wmma_bf16(const bf16_t* __restrict__ A,
                               const bf16_t* __restrict__ Bt,
                               const float* __restrict__ Cin,
                               float* __restrict__ Of32, int ldc,
                               bf16_t* __restrict__ ObfT, int ldt,
                               int K, int act) {
  const int lane = threadIdx.x & 31;
  const int l15  = lane & 15;
  const int hi   = lane >> 4;
  const int n0   = blockIdx.x * (16 * NT);
  const int m0   = blockIdx.y * (16 * MT);

  v8f c[MT][NT];
  if (Cin) {
#pragma unroll
    for (int mt = 0; mt < MT; ++mt)
#pragma unroll
      for (int nt = 0; nt < NT; ++nt)
#pragma unroll
        for (int r = 0; r < 8; ++r)
          c[mt][nt][r] = Cin[(size_t)(m0 + mt * 16 + 8 * hi + r) * ldc + n0 + nt * 16 + l15];
  } else {
#pragma unroll
    for (int mt = 0; mt < MT; ++mt)
#pragma unroll
      for (int nt = 0; nt < NT; ++nt)
#pragma unroll
        for (int r = 0; r < 8; ++r) c[mt][nt][r] = 0.f;
  }

  // A fragment (16-bit A 16x32): lane row = l15; K = hi*8+0..7 and 16+hi*8+0..7.
  // B fragment (16-bit B 32x16): lane col = l15; K = hi*8+0..7 and 16+hi*8+0..7 (same as A).
  const bf16_t* abase = A  + (size_t)(m0 + l15) * LD;
  const bf16_t* bbase = Bt + (size_t)(n0 + l15) * LD;

  for (int k = 0; k < K; k += 32) {
    union Frag { v16bf v; v8bf h[2]; } a[MT], b[NT];
#pragma unroll
    for (int mt = 0; mt < MT; ++mt) {
      a[mt].h[0] = *reinterpret_cast<const v8bf*>(abase + (size_t)mt * 16 * LD + k + 8 * hi);
      a[mt].h[1] = *reinterpret_cast<const v8bf*>(abase + (size_t)mt * 16 * LD + k + 16 + 8 * hi);
      __builtin_prefetch(abase + (size_t)mt * 16 * LD + k + 256, 0, 3);  // stream A ahead
    }
#pragma unroll
    for (int nt = 0; nt < NT; ++nt) {
      b[nt].h[0] = *reinterpret_cast<const v8bf*>(bbase + (size_t)nt * 16 * LD + k + 8 * hi);
      b[nt].h[1] = *reinterpret_cast<const v8bf*>(bbase + (size_t)nt * 16 * LD + k + 16 + 8 * hi);
    }
    // 8 independent WMMAs per k-step: maximal XDL issue, operands reused from VGPRs
#pragma unroll
    for (int mt = 0; mt < MT; ++mt)
#pragma unroll
      for (int nt = 0; nt < NT; ++nt)
        c[mt][nt] = __builtin_amdgcn_wmma_f32_16x16x32_bf16(
            false, a[mt].v, false, b[nt].v, (short)0, c[mt][nt], false, false);
  }

  // C/D layout: value (vgpr r, lane) = C[m0+mt*16 + r + 8*hi, n0+nt*16 + l15]
  if (Of32) {
#pragma unroll
    for (int mt = 0; mt < MT; ++mt)
#pragma unroll
      for (int nt = 0; nt < NT; ++nt)
#pragma unroll
        for (int r = 0; r < 8; ++r) {
          float v = c[mt][nt][r];
          if (act == 2) v = v > 0.f ? v : expm1f(v);   // ELU
          Of32[(size_t)(m0 + mt * 16 + 8 * hi + r) * ldc + n0 + nt * 16 + l15] = v;
        }
  }
  if (ObfT) {  // store D^T bf16: per-lane 8 contiguous along M -> one 16B store per tile
#pragma unroll
    for (int mt = 0; mt < MT; ++mt)
#pragma unroll
      for (int nt = 0; nt < NT; ++nt) {
        v8bf pk;
#pragma unroll
        for (int r = 0; r < 8; ++r) pk[r] = f2b(c[mt][nt][r]);
        *reinterpret_cast<v8bf*>(ObfT + (size_t)(n0 + nt * 16 + l15) * ldt
                                 + m0 + mt * 16 + 8 * hi) = pk;
      }
  }
}

// ---------------- row dot: z[r] = X[r,:128] . a ----------------
__global__ void rowdot128(const float* __restrict__ X, const float* __restrict__ a,
                          float* __restrict__ z) {
  __shared__ float sh[128];
  const int r = blockIdx.x, tid = threadIdx.x;
  sh[tid] = X[(size_t)r * F_OUT + tid] * a[tid];
  __syncthreads();
  for (int s = 64; s > 0; s >>= 1) {
    if (tid < s) sh[tid] += sh[tid + s];
    __syncthreads();
  }
  if (tid == 0) z[r] = sh[0];
}

// ---------------- masked softmax over rows of e12; writes att and att^T ----------------
__global__ void att_softmax(const int* __restrict__ adj, const float* __restrict__ z2,
                            const float* __restrict__ z1, bf16_t* __restrict__ att,
                            bf16_t* __restrict__ attT) {
  __shared__ float wbuf[M_NODES];
  __shared__ float red[256];
  const int tid = threadIdx.x;
  const int n = blockIdx.x;
  const float zn = z2[n];
  const int* arow = adj + (size_t)n * M_NODES;
  float mx = -INFINITY;
  for (int m = tid; m < M_NODES; m += 256) {
    float e = zn + z1[m];
    e = e > 0.f ? e : 0.2f * e;                 // leaky(0.2)
    float w = arow[m] > 0 ? e : NEGV;
    wbuf[m] = w;
    mx = fmaxf(mx, w);
  }
  red[tid] = mx; __syncthreads();
  for (int s = 128; s > 0; s >>= 1) {
    if (tid < s) red[tid] = fmaxf(red[tid], red[tid + s]);
    __syncthreads();
  }
  const float rowmax = red[0];
  __syncthreads();
  float sum = 0.f;
  for (int m = tid; m < M_NODES; m += 256) {
    float p = __expf(wbuf[m] - rowmax);         // masked entries underflow to 0
    wbuf[m] = p; sum += p;
  }
  red[tid] = sum; __syncthreads();
  for (int s = 128; s > 0; s >>= 1) {
    if (tid < s) red[tid] += red[tid + s];
    __syncthreads();
  }
  const float inv = 1.f / red[0];
  for (int m = tid; m < M_NODES; m += 256) {
    bf16_t p = __float2bfloat16(wbuf[m] * inv);
    att [(size_t)n * M_NODES + m] = p;
    attT[(size_t)m * N_NODES + n] = p;
  }
}

// ---------------- adjacency row counts -> 1/cnt (negative flag => empty row) ----------------
__global__ void row_count(const int* __restrict__ adjC, const int* __restrict__ adjP,
                          const int* __restrict__ src, float* __restrict__ invC,
                          float* __restrict__ invP) {
  __shared__ int sc[256], sp[256];
  const int b = blockIdx.x, tid = threadIdx.x;
  const int s = src[b];
  const int* rc = adjC + (size_t)s * N_NODES;
  const int* rp = adjP + (size_t)s * N_NODES;
  int cc = 0, cp = 0;
  for (int i = tid; i < N_NODES; i += 256) { cc += (rc[i] > 0); cp += (rp[i] > 0); }
  sc[tid] = cc; sp[tid] = cp; __syncthreads();
  for (int t = 128; t > 0; t >>= 1) {
    if (tid < t) { sc[tid] += sc[tid + t]; sp[tid] += sp[tid + t]; }
    __syncthreads();
  }
  if (tid == 0) {
    invC[b] = sc[0] ? 1.f / (float)sc[0] : -1.f / (float)N_NODES;
    invP[b] = sp[0] ? 1.f / (float)sp[0] : -1.f / (float)N_NODES;
  }
}

// Ct[n,b] = att3^T + att4^T combined coefficients (row-constant softmax == uniform on support)
__global__ void build_ct(const int* __restrict__ adjC, const int* __restrict__ adjP,
                         const int* __restrict__ src, const float* __restrict__ invC,
                         const float* __restrict__ invP, bf16_t* __restrict__ Ct) {
  const int n = blockIdx.x * 256 + threadIdx.x;
  const int b = blockIdx.y;
  const int s = src[b];
  const float ic = invC[b], ip = invP[b];
  float vc = (ic < 0.f) ? -ic : (adjC[(size_t)s * N_NODES + n] > 0 ? ic : 0.f);
  float vp = (ip < 0.f) ? -ip : (adjP[(size_t)s * N_NODES + n] > 0 ? ip : 0.f);
  Ct[(size_t)n * B_SRC + b] = __float2bfloat16(vc + vp);
}

// h2selT[f,b] = h2[src[b], f]  (B^T operand for the Ct GEMM)
__global__ void gather_h2selT(const float* __restrict__ h2f, const int* __restrict__ src,
                              bf16_t* __restrict__ outT) {
  const int b = blockIdx.x, f = threadIdx.x;
  outT[(size_t)f * B_SRC + b] = __float2bfloat16(h2f[(size_t)src[b] * F_OUT + f]);
}

// ---------------- batchnorm (training stats) ----------------
__global__ void bn_stats(const float* __restrict__ X, int rows,
                         float* __restrict__ sum, float* __restrict__ sumsq) {
  const int f = threadIdx.x;            // 128 threads
  const int r0 = blockIdx.x * 64;
  float s = 0.f, q = 0.f;
  for (int r = r0; r < r0 + 64; ++r) {
    float v = X[(size_t)r * F_OUT + f];
    s += v; q += v * v;
  }
  atomicAdd(&sum[f], s);
  atomicAdd(&sumsq[f], q);
}

__global__ void bn_apply(const float* __restrict__ X, const float* __restrict__ sum,
                         const float* __restrict__ sumsq, const float* __restrict__ gamma,
                         const float* __restrict__ beta, int rows, bf16_t* __restrict__ out) {
  const int i = blockIdx.x * 256 + threadIdx.x;
  if (i >= rows * F_OUT) return;
  const int f = i & (F_OUT - 1);
  const float inv_n = 1.f / (float)rows;
  const float mean = sum[f] * inv_n;
  const float var  = sumsq[f] * inv_n - mean * mean;
  float y = (X[i] - mean) * rsqrtf(var + 1e-5f) * gamma[f] + beta[f];
  y = y > 0.f ? y : 0.2f * y;           // leaky
  out[i] = __float2bfloat16(y);
}

// ---------------- host-side orchestration ----------------
extern "C" void kernel_launch(void* const* d_in, const int* in_sizes, int n_in,
                              void* d_out, int out_size, void* d_ws, size_t ws_size,
                              hipStream_t stream) {
  (void)in_sizes; (void)n_in; (void)out_size; (void)ws_size;
  const float* Sinput    = (const float*)d_in[0];
  const float* Rinput    = (const float*)d_in[1];
  const int*   inter_adj = (const int*)d_in[2];
  const int*   city_adj  = (const int*)d_in[3];
  const int*   prov_adj  = (const int*)d_in[4];
  const int*   src       = (const int*)d_in[5];
  const float* W1        = (const float*)d_in[6];
  const float* W2        = (const float*)d_in[7];
  const float* a_vec     = (const float*)d_in[8];
  // d_in[9] (a3) and d_in[10] (a4) cancel: softmax of a row-constant score over a
  // mask is uniform on the mask support (exp(NEG - c) underflows to 0 in fp32).
  const float* bn1_g = (const float*)d_in[11];
  const float* bn1_b = (const float*)d_in[12];
  const float* bn2_g = (const float*)d_in[13];
  const float* bn2_b = (const float*)d_in[14];
  float* out = (float*)d_out;

  size_t off = 0;
  auto alloc = [&](size_t bytes) -> void* {
    off = (off + 255) & ~(size_t)255;
    void* p = (char*)d_ws + off;
    off += bytes;
    return p;
  };

  bf16_t* Sb      = (bf16_t*)alloc((size_t)N_NODES * F_IN * 2);
  bf16_t* Rb      = (bf16_t*)alloc((size_t)M_NODES * F_IN * 2);
  bf16_t* Wt1     = (bf16_t*)alloc((size_t)F_OUT * F_IN * 2);
  bf16_t* Wt2     = (bf16_t*)alloc((size_t)F_OUT * F_IN * 2);
  float*  h2f     = (float*) alloc((size_t)N_NODES * F_OUT * 4);
  float*  h1f     = (float*) alloc((size_t)M_NODES * F_OUT * 4);
  bf16_t* h2T     = (bf16_t*)alloc((size_t)F_OUT * N_NODES * 2);
  bf16_t* h1T     = (bf16_t*)alloc((size_t)F_OUT * M_NODES * 2);
  float*  z2      = (float*) alloc((size_t)N_NODES * 4);
  float*  z1      = (float*) alloc((size_t)M_NODES * 4);
  bf16_t* att     = (bf16_t*)alloc((size_t)N_NODES * M_NODES * 2);
  bf16_t* attT    = (bf16_t*)alloc((size_t)M_NODES * N_NODES * 2);
  float*  invC    = (float*) alloc((size_t)B_SRC * 4);
  float*  invP    = (float*) alloc((size_t)B_SRC * 4);
  bf16_t* Ct      = (bf16_t*)alloc((size_t)N_NODES * B_SRC * 2);
  bf16_t* h2selT  = (bf16_t*)alloc((size_t)F_OUT * B_SRC * 2);
  float*  InterRC = (float*) alloc((size_t)N_NODES * F_OUT * 4);
  float*  Upre    = (float*) alloc((size_t)N_NODES * F_OUT * 4);
  float*  Vpre    = (float*) alloc((size_t)M_NODES * F_OUT * 4);
  float*  bnbuf   = (float*) alloc((size_t)4 * F_OUT * 4);
  bf16_t* Ub      = (bf16_t*)alloc((size_t)N_NODES * F_OUT * 2);
  bf16_t* Vb      = (bf16_t*)alloc((size_t)M_NODES * F_OUT * 2);

  const int TN = 16 * NT;   // 64 columns per wave
  const int TM = 16 * MT;   // 32 rows per wave

  // 1) casts / weight transposes
  cast_f32_bf16<<<(N_NODES * F_IN + 255) / 256, 256, 0, stream>>>(Sinput, Sb, N_NODES * F_IN);
  cast_f32_bf16<<<(M_NODES * F_IN + 255) / 256, 256, 0, stream>>>(Rinput, Rb, M_NODES * F_IN);
  transpose_cast<<<(F_IN * F_OUT + 255) / 256, 256, 0, stream>>>(W1, Wt1, F_IN, F_OUT);
  transpose_cast<<<(F_IN * F_OUT + 255) / 256, 256, 0, stream>>>(W2, Wt2, F_IN, F_OUT);

  // 2) h2 = S@W2, h1 = R@W1 (also emit transposed bf16 copies for later GEMMs)
  gemm_wmma_bf16<F_IN><<<dim3(F_OUT / TN, N_NODES / TM), 32, 0, stream>>>(
      Sb, Wt2, nullptr, h2f, F_OUT, h2T, N_NODES, F_IN, 0);
  gemm_wmma_bf16<F_IN><<<dim3(F_OUT / TN, M_NODES / TM), 32, 0, stream>>>(
      Rb, Wt1, nullptr, h1f, F_OUT, h1T, M_NODES, F_IN, 0);

  // 3) attention scores (rank-1 decomposition) + masked softmax
  rowdot128<<<N_NODES, 128, 0, stream>>>(h2f, a_vec + F_OUT, z2);
  rowdot128<<<M_NODES, 128, 0, stream>>>(h1f, a_vec, z1);
  att_softmax<<<N_NODES, 256, 0, stream>>>(inter_adj, z2, z1, att, attT);

  // 4) IntraNC coefficient matrix (uniform softmax over adjacency support)
  row_count<<<B_SRC, 256, 0, stream>>>(city_adj, prov_adj, src, invC, invP);
  build_ct<<<dim3(N_NODES / 256, B_SRC), 256, 0, stream>>>(city_adj, prov_adj, src, invC, invP, Ct);
  gather_h2selT<<<B_SRC, F_OUT, 0, stream>>>(h2f, src, h2selT);

  // 5) big WMMA GEMMs
  gemm_wmma_bf16<M_NODES><<<dim3(F_OUT / TN, N_NODES / TM), 32, 0, stream>>>(
      att, h1T, nullptr, InterRC, F_OUT, nullptr, 0, M_NODES, 0);                 // InterRC
  gemm_wmma_bf16<B_SRC><<<dim3(F_OUT / TN, N_NODES / TM), 32, 0, stream>>>(
      Ct, h2selT, InterRC, Upre, F_OUT, nullptr, 0, B_SRC, 0);                    // + IntraNC
  gemm_wmma_bf16<N_NODES><<<dim3(F_OUT / TN, M_NODES / TM), 32, 0, stream>>>(
      attT, h2T, nullptr, Vpre, F_OUT, nullptr, 0, N_NODES, 0);                   // att^T @ h2

  // 6) batchnorm + leaky -> bf16 operands for the final GEMM
  zero_f32<<<(4 * F_OUT + 255) / 256, 256, 0, stream>>>(bnbuf, 4 * F_OUT);
  bn_stats<<<N_NODES / 64, 128, 0, stream>>>(Upre, N_NODES, bnbuf + 0 * F_OUT, bnbuf + 1 * F_OUT);
  bn_stats<<<M_NODES / 64, 128, 0, stream>>>(Vpre, M_NODES, bnbuf + 2 * F_OUT, bnbuf + 3 * F_OUT);
  bn_apply<<<(N_NODES * F_OUT + 255) / 256, 256, 0, stream>>>(
      Upre, bnbuf + 0 * F_OUT, bnbuf + 1 * F_OUT, bn2_g, bn2_b, N_NODES, Ub);
  bn_apply<<<(M_NODES * F_OUT + 255) / 256, 256, 0, stream>>>(
      Vpre, bnbuf + 2 * F_OUT, bnbuf + 3 * F_OUT, bn1_g, bn1_b, M_NODES, Vb);

  // 7) h_prime = u @ v^T with fused ELU -> d_out (Vb row-major [M,F] == B^T layout)
  gemm_wmma_bf16<F_OUT><<<dim3(M_NODES / TN, N_NODES / TM), 32, 0, stream>>>(
      Ub, Vb, nullptr, out, M_NODES, nullptr, 0, F_OUT, 2);
}
